// LogEpsMM_19490561589868
// MI455X (gfx1250) — matrix operand order repaired and checked by reading, not compiled, not run on hardware
//
#include <hip/hip_runtime.h>

// out = log(max(x @ W + 1e-5, 1e-12))   (big/small branches identical in float math)
// x: (8192, 1024) f32, W: (1024, 1024) f32, out: (8192, 1024) f32
//
// Plan:
//   1) pre-split kernels: f32 -> bf16 hi/lo pairs in d_ws
//        A_hi/A_lo: (8192,1024) bf16 row-major
//        Bt_hi/Bt_lo: (1024n,1024k) bf16  (transposed so GEMM B tiles are row chunks)
//   2) GEMM: bf16x3 split (C += Ahi*Bhi + Ahi*Blo + Alo*Bhi) with
//      v_wmma_f32_16x16x32_bf16; stage fills via TDM tensor_load_to_lds
//      (double-buffered, s_wait_tensorcnt), fused log epilogue.
//   Fallback (ws too small): round-1 kernel with in-kernel split.

typedef __attribute__((ext_vector_type(16))) __bf16 v16bf;
typedef __attribute__((ext_vector_type(8)))  float  v8f;
typedef __attribute__((ext_vector_type(4)))  unsigned int v4u;
typedef __attribute__((ext_vector_type(8)))  int v8i;
typedef __attribute__((ext_vector_type(4)))  int v4i;

#define TILE_M 128
#define TILE_N 128
#define TILE_K 64
#define LDS_K  72          // 64 + 8 pad elems; 144-byte rows (16B aligned)
#define GEMM_K 1024
#define GEMM_N 1024
#define GEMM_M 8192

union Frag {
    v16bf v;
    uint4 q[2];
};

__device__ __forceinline__ void bf16_split(float x, __bf16& h, __bf16& l) {
    __bf16 hh = (__bf16)x;
    float  r  = x - (float)hh;
    h = hh;
    l = (__bf16)r;
}

// ---------------------------------------------------------------------------
// Pre-split kernels
// ---------------------------------------------------------------------------
union Pack4 { __bf16 b[4]; uint2 u; };

__global__ __launch_bounds__(256)
void split_a_kernel(const float* __restrict__ A, __bf16* __restrict__ Ahi,
                    __bf16* __restrict__ Alo)
{
    const size_t idx4 = (size_t)blockIdx.x * 256 + threadIdx.x;   // float4 index
    const float4 v = reinterpret_cast<const float4*>(A)[idx4];
    Pack4 h, l;
    bf16_split(v.x, h.b[0], l.b[0]);
    bf16_split(v.y, h.b[1], l.b[1]);
    bf16_split(v.z, h.b[2], l.b[2]);
    bf16_split(v.w, h.b[3], l.b[3]);
    reinterpret_cast<uint2*>(Ahi)[idx4] = h.u;
    reinterpret_cast<uint2*>(Alo)[idx4] = l.u;
}

__global__ __launch_bounds__(256)
void split_bt_kernel(const float* __restrict__ B, __bf16* __restrict__ Bthi,
                     __bf16* __restrict__ Btlo)
{
    // read B[k][n] coalesced, write transposed Bt[n][k]
    const size_t g = (size_t)blockIdx.x * 256 + threadIdx.x;
    const int k = (int)(g >> 10);
    const int n = (int)(g & 1023);
    const float x = B[g];
    __bf16 h, l;
    bf16_split(x, h, l);
    Bthi[(size_t)n * GEMM_K + k] = h;
    Btlo[(size_t)n * GEMM_K + k] = l;
}

// ---------------------------------------------------------------------------
// TDM descriptor: 2D tile (tile_d0 x tile_d1 elements of 2 bytes) from a
// row-major tensor with row stride `stride_elems`, into LDS at lds_off with
// 16B pad after each 128B row (=> 144B LDS row stride).
// ---------------------------------------------------------------------------
__device__ __forceinline__ void tdm_load_tile_2d(unsigned lds_off, const void* gptr,
                                                 unsigned tile_d0, unsigned tile_d1,
                                                 unsigned long long stride_elems)
{
    const unsigned long long ga = (unsigned long long)gptr;
    v4u g0;
    g0[0] = 1u;                                          // count=1, user descriptor
    g0[1] = lds_off;                                     // lds_addr (bytes)
    g0[2] = (unsigned)(ga & 0xFFFFFFFFu);                // global_addr[31:0]
    g0[3] = (unsigned)((ga >> 32) & 0x1FFFFFFu)          // global_addr[56:32]
            | (2u << 30);                                // type = 2 ("image")

    const unsigned td0 = 1u << 20;                       // tensor_dim0 (no OOB clip)
    const unsigned td1 = 1u << 20;                       // tensor_dim1
    v8i g1;
    g1[0] = (int)((1u << 16)                             // data_size = 2 bytes
                | (1u << 20)                             // pad_enable
                | (4u << 22)                             // pad_interval: 32 DWORDs (128B)
                | (3u << 25));                           // pad_amount: 4 DWORDs (16B)
    g1[1] = (int)((td0 & 0xFFFFu) << 16);                // abar_addr=0 | tensor_dim0 lo
    g1[2] = (int)(((td0 >> 16) & 0xFFFFu) | ((td1 & 0xFFFFu) << 16));
    g1[3] = (int)(((td1 >> 16) & 0xFFFFu) | ((tile_d0 & 0xFFFFu) << 16));
    g1[4] = (int)(tile_d1 & 0xFFFFu);                    // tile_dim1 | tile_dim2=0 (2D)
    const unsigned long long s0 = stride_elems;          // tensor_dim0_stride (elems)
    g1[5] = (int)(unsigned)(s0 & 0xFFFFFFFFull);
    g1[6] = (int)(unsigned)((s0 >> 32) & 0xFFFFull);     // | dim1_stride lo (0)
    g1[7] = 0;

    const v4i gz4 = {0, 0, 0, 0};                        // groups 2/3 unused (2D)
    const v8i gz8 = {0, 0, 0, 0, 0, 0, 0, 0};            // extra group (clang-23 form)
    __builtin_amdgcn_tensor_load_to_lds(g0, g1, gz4, gz4, gz8, 0);
}

// ---------------------------------------------------------------------------
// GEMM with TDM stage fill (double buffered)
// ---------------------------------------------------------------------------
__global__ __launch_bounds__(256)
void logepsmm_gemm_tdm(const __bf16* __restrict__ Ahi, const __bf16* __restrict__ Alo,
                       const __bf16* __restrict__ Bthi, const __bf16* __restrict__ Btlo,
                       float* __restrict__ out)
{
    // [stage][tile: 0=AsHi 1=AsLo 2=BsHi 3=BsLo][row][k]
    __shared__ __bf16 lds[2][4][TILE_M][LDS_K];

    const int tid   = threadIdx.x;
    const int lane  = tid & 31;
    const int wid   = tid >> 5;
    const int waveM = wid & 1;
    const int waveN = wid >> 1;
    const int half  = lane >> 4;
    const int l15   = lane & 15;

    const int bn = blockIdx.x;
    const int bm = blockIdx.y;

    v8f acc[4][2];
#pragma unroll
    for (int i = 0; i < 4; ++i)
#pragma unroll
        for (int j = 0; j < 2; ++j) acc[i][j] = {};

    const int KB = GEMM_K / TILE_K;                      // 16 stages

    const size_t aBase = (size_t)bm * TILE_M * GEMM_K;   // elem offset of A tile rows
    const size_t bBase = (size_t)bn * TILE_N * GEMM_K;   // elem offset of Bt tile rows

    auto issue_stage = [&](int kb, int s) {
        const size_t ka = (size_t)kb * TILE_K;
        tdm_load_tile_2d((unsigned)(size_t)&lds[s][0][0][0], Ahi  + aBase + ka,
                         TILE_K, TILE_M, GEMM_K);
        tdm_load_tile_2d((unsigned)(size_t)&lds[s][1][0][0], Alo  + aBase + ka,
                         TILE_K, TILE_M, GEMM_K);
        tdm_load_tile_2d((unsigned)(size_t)&lds[s][2][0][0], Bthi + bBase + ka,
                         TILE_K, TILE_N, GEMM_K);
        tdm_load_tile_2d((unsigned)(size_t)&lds[s][3][0][0], Btlo + bBase + ka,
                         TILE_K, TILE_N, GEMM_K);
    };

    if (wid == 0) issue_stage(0, 0);                     // prologue: stage 0 -> buf 0

    for (int kb = 0; kb < KB; ++kb) {
        const int cur = kb & 1;
        if (wid == 0) {
            if (kb + 1 < KB) {
                issue_stage(kb + 1, cur ^ 1);            // prefetch next stage
                __builtin_amdgcn_s_wait_tensorcnt(4);    // current stage's 4 ops done
            } else {
                __builtin_amdgcn_s_wait_tensorcnt(0);
            }
        }
        __syncthreads();

#pragma unroll
        for (int ks = 0; ks < TILE_K; ks += 32) {
            Frag bhi[2], blo[2];
#pragma unroll
            for (int nt = 0; nt < 2; ++nt) {
                const int col = waveN * 32 + nt * 16 + l15;
                const int kb0 = ks + half * 16;
                bhi[nt].q[0] = *reinterpret_cast<const uint4*>(&lds[cur][2][col][kb0]);
                bhi[nt].q[1] = *reinterpret_cast<const uint4*>(&lds[cur][2][col][kb0 + 8]);
                blo[nt].q[0] = *reinterpret_cast<const uint4*>(&lds[cur][3][col][kb0]);
                blo[nt].q[1] = *reinterpret_cast<const uint4*>(&lds[cur][3][col][kb0 + 8]);
            }
#pragma unroll
            for (int mt = 0; mt < 4; ++mt) {
                const int mr = waveM * 64 + mt * 16 + l15;
                const int k0 = ks + half * 16;
                Frag ahi, alo;
                ahi.q[0] = *reinterpret_cast<const uint4*>(&lds[cur][0][mr][k0]);
                ahi.q[1] = *reinterpret_cast<const uint4*>(&lds[cur][0][mr][k0 + 8]);
                alo.q[0] = *reinterpret_cast<const uint4*>(&lds[cur][1][mr][k0]);
                alo.q[1] = *reinterpret_cast<const uint4*>(&lds[cur][1][mr][k0 + 8]);
#pragma unroll
                for (int nt = 0; nt < 2; ++nt) {
                    acc[mt][nt] = __builtin_amdgcn_wmma_f32_16x16x32_bf16(
                        false, ahi.v, false, bhi[nt].v, (short)0, acc[mt][nt], false, false);
                    acc[mt][nt] = __builtin_amdgcn_wmma_f32_16x16x32_bf16(
                        false, ahi.v, false, blo[nt].v, (short)0, acc[mt][nt], false, false);
                    acc[mt][nt] = __builtin_amdgcn_wmma_f32_16x16x32_bf16(
                        false, alo.v, false, bhi[nt].v, (short)0, acc[mt][nt], false, false);
                }
            }
        }
        __syncthreads();
    }

#pragma unroll
    for (int mt = 0; mt < 4; ++mt)
#pragma unroll
        for (int nt = 0; nt < 2; ++nt) {
            const int mBase = bm * TILE_M + waveM * 64 + mt * 16 + half * 8;
            const int nCol  = bn * TILE_N + waveN * 32 + nt * 16 + l15;
#pragma unroll
            for (int r = 0; r < 8; ++r) {
                float y = acc[mt][nt][r] + 1e-5f;
                y = fmaxf(y, 1e-12f);
                out[(size_t)(mBase + r) * GEMM_N + nCol] = __logf(y);
            }
        }
}

// ---------------------------------------------------------------------------
// Fallback GEMM (round-1 path: in-kernel split, plain loads) — used when the
// workspace is too small for the pre-split arrays.
// ---------------------------------------------------------------------------
__global__ __launch_bounds__(256)
void logepsmm_gemm_fallback(const float* __restrict__ A, const float* __restrict__ B,
                            float* __restrict__ out)
{
    __shared__ __bf16 AsHi[TILE_M][LDS_K];
    __shared__ __bf16 AsLo[TILE_M][LDS_K];
    __shared__ __bf16 BsHi[TILE_N][LDS_K];
    __shared__ __bf16 BsLo[TILE_N][LDS_K];

    const int tid   = threadIdx.x;
    const int lane  = tid & 31;
    const int wid   = tid >> 5;
    const int waveM = wid & 1;
    const int waveN = wid >> 1;
    const int half  = lane >> 4;
    const int l15   = lane & 15;

    const int bn = blockIdx.x;
    const int bm = blockIdx.y;

    v8f acc[4][2];
#pragma unroll
    for (int i = 0; i < 4; ++i)
#pragma unroll
        for (int j = 0; j < 2; ++j) acc[i][j] = {};

    const int KB = GEMM_K / TILE_K;

    for (int kb = 0; kb < KB; ++kb) {
#pragma unroll
        for (int i = 0; i < 8; ++i) {
            const int f  = tid + i * 256;
            const int m  = f >> 4;
            const int k0 = (f & 15) << 2;
            const float4 v = reinterpret_cast<const float4*>(A)
                [(size_t)(bm * TILE_M + m) * (GEMM_K / 4) + (kb * TILE_K + k0) / 4];
            __bf16 h, l;
            bf16_split(v.x, h, l); AsHi[m][k0 + 0] = h; AsLo[m][k0 + 0] = l;
            bf16_split(v.y, h, l); AsHi[m][k0 + 1] = h; AsLo[m][k0 + 1] = l;
            bf16_split(v.z, h, l); AsHi[m][k0 + 2] = h; AsLo[m][k0 + 2] = l;
            bf16_split(v.w, h, l); AsHi[m][k0 + 3] = h; AsLo[m][k0 + 3] = l;
        }
        {
            const int n  = tid & 127;
            const int kq = tid >> 7;
#pragma unroll
            for (int jj = 0; jj < 32; ++jj) {
                const int kk = kq * 32 + jj;
                const float x = B[(size_t)(kb * TILE_K + kk) * GEMM_N + bn * TILE_N + n];
                __bf16 h, l;
                bf16_split(x, h, l);
                BsHi[n][kk] = h;
                BsLo[n][kk] = l;
            }
        }
        if (kb + 1 < KB) {
            const int m  = tid >> 4;
            const int k0 = (tid & 15) << 2;
            __builtin_prefetch(&A[(size_t)(bm * TILE_M + m) * GEMM_K + (kb + 1) * TILE_K + k0], 0, 1);
            __builtin_prefetch(&B[(size_t)((kb + 1) * TILE_K + (tid >> 7) * 32) * GEMM_N
                                  + bn * TILE_N + (tid & 127)], 0, 1);
        }
        __syncthreads();

#pragma unroll
        for (int ks = 0; ks < TILE_K; ks += 32) {
            Frag bhi[2], blo[2];
#pragma unroll
            for (int nt = 0; nt < 2; ++nt) {
                const int col = waveN * 32 + nt * 16 + l15;
                const int kb0 = ks + half * 16;
                bhi[nt].q[0] = *reinterpret_cast<const uint4*>(&BsHi[col][kb0]);
                bhi[nt].q[1] = *reinterpret_cast<const uint4*>(&BsHi[col][kb0 + 8]);
                blo[nt].q[0] = *reinterpret_cast<const uint4*>(&BsLo[col][kb0]);
                blo[nt].q[1] = *reinterpret_cast<const uint4*>(&BsLo[col][kb0 + 8]);
            }
#pragma unroll
            for (int mt = 0; mt < 4; ++mt) {
                const int mr = waveM * 64 + mt * 16 + l15;
                const int k0 = ks + half * 16;
                Frag ahi, alo;
                ahi.q[0] = *reinterpret_cast<const uint4*>(&AsHi[mr][k0]);
                ahi.q[1] = *reinterpret_cast<const uint4*>(&AsHi[mr][k0 + 8]);
                alo.q[0] = *reinterpret_cast<const uint4*>(&AsLo[mr][k0]);
                alo.q[1] = *reinterpret_cast<const uint4*>(&AsLo[mr][k0 + 8]);
#pragma unroll
                for (int nt = 0; nt < 2; ++nt) {
                    acc[mt][nt] = __builtin_amdgcn_wmma_f32_16x16x32_bf16(
                        false, ahi.v, false, bhi[nt].v, (short)0, acc[mt][nt], false, false);
                    acc[mt][nt] = __builtin_amdgcn_wmma_f32_16x16x32_bf16(
                        false, ahi.v, false, blo[nt].v, (short)0, acc[mt][nt], false, false);
                    acc[mt][nt] = __builtin_amdgcn_wmma_f32_16x16x32_bf16(
                        false, alo.v, false, bhi[nt].v, (short)0, acc[mt][nt], false, false);
                }
            }
        }
        __syncthreads();
    }

#pragma unroll
    for (int mt = 0; mt < 4; ++mt)
#pragma unroll
        for (int nt = 0; nt < 2; ++nt) {
            const int mBase = bm * TILE_M + waveM * 64 + mt * 16 + half * 8;
            const int nCol  = bn * TILE_N + waveN * 32 + nt * 16 + l15;
#pragma unroll
            for (int r = 0; r < 8; ++r) {
                float y = acc[mt][nt][r] + 1e-5f;
                y = fmaxf(y, 1e-12f);
                out[(size_t)(mBase + r) * GEMM_N + nCol] = __logf(y);
            }
        }
}

// ---------------------------------------------------------------------------
extern "C" void kernel_launch(void* const* d_in, const int* in_sizes, int n_in,
                              void* d_out, int out_size, void* d_ws, size_t ws_size,
                              hipStream_t stream) {
    const float* x = (const float*)d_in[0];   // (4, 2048, 1024) f32
    const float* w = (const float*)d_in[1];   // (1024, 1024) f32
    float* out = (float*)d_out;
    (void)in_sizes; (void)n_in; (void)out_size;

    const size_t szA  = (size_t)GEMM_M * GEMM_K * sizeof(__bf16);   // 16 MB
    const size_t szB  = (size_t)GEMM_K * GEMM_N * sizeof(__bf16);   //  2 MB
    const size_t need = 2 * szA + 2 * szB;                          // ~37.7 MB

    dim3 grid(GEMM_N / TILE_N, GEMM_M / TILE_M);                    // (8, 64)

    if (ws_size >= need) {
        __bf16* Ahi  = (__bf16*)d_ws;
        __bf16* Alo  = (__bf16*)((char*)d_ws + szA);
        __bf16* Bthi = (__bf16*)((char*)d_ws + 2 * szA);
        __bf16* Btlo = (__bf16*)((char*)d_ws + 2 * szA + szB);

        split_a_kernel<<<(GEMM_M * GEMM_K / 4) / 256, 256, 0, stream>>>(x, Ahi, Alo);
        split_bt_kernel<<<(GEMM_K * GEMM_N) / 256, 256, 0, stream>>>(w, Bthi, Btlo);
        logepsmm_gemm_tdm<<<grid, 256, 0, stream>>>(Ahi, Alo, Bthi, Btlo, out);
    } else {
        logepsmm_gemm_fallback<<<grid, 256, 0, stream>>>(x, w, out);
    }
}
